// GCNSubgraphAdaptiveExit_87608742904008
// MI455X (gfx1250) — matrix operand order repaired and checked by reading, not compiled, not run on hardware
//
#include <hip/hip_runtime.h>

#define NN 50000
#define EE 800000
#define HH 128
#define CH 64
#define LL 3
#define LDS_STRIDE 65   // 64 + 1 pad: avoid 64-bank conflicts in stage-2 dots

typedef __attribute__((ext_vector_type(16))) _Float16 v16h;
typedef __attribute__((ext_vector_type(8)))  float    v8f;

// ---------------------------------------------------------------------------
// Init: zero z (in d_out), reset per-node state, deg=1 (self loop), active=0
// ---------------------------------------------------------------------------
__global__ __launch_bounds__(256) void init_all(float* __restrict__ z,
                                                float* __restrict__ soft_ex,
                                                int* __restrict__ hard_ex,
                                                int* __restrict__ exit_l,
                                                int* __restrict__ active,
                                                float* __restrict__ deg,
                                                float* __restrict__ deg_ns) {
  long t = (long)blockIdx.x * 256 + threadIdx.x;
  if (t < (long)NN * HH) z[t] = 0.0f;
  if (t < NN) {
    soft_ex[t] = 0.0f;
    hard_ex[t] = 0;
    exit_l[t]  = LL;
    deg[t]     = 1.0f;   // self-loop contribution to in-degree
    deg_ns[t]  = 0.0f;
  }
  if (t < LL) active[t] = 0;
}

// ---------------------------------------------------------------------------
// Degree accumulation over edges (dst for GCN norm, src non-self for readiness)
// ---------------------------------------------------------------------------
__global__ __launch_bounds__(256) void deg_edges(const int* __restrict__ src,
                                                 const int* __restrict__ dst,
                                                 float* __restrict__ deg,
                                                 float* __restrict__ deg_ns) {
  int e = blockIdx.x * 256 + threadIdx.x;
  if (e < EE) {
    int s = src[e], d = dst[e];
    unsafeAtomicAdd(&deg[d], 1.0f);
    if (s != d) unsafeAtomicAdd(&deg_ns[s], 1.0f);
  }
}

__global__ __launch_bounds__(256) void dinv_fix(const float* __restrict__ deg,
                                                float* __restrict__ dinv,
                                                float* __restrict__ deg_ns) {
  int n = blockIdx.x * 256 + threadIdx.x;
  if (n < NN) {
    dinv[n]   = rsqrtf(deg[n]);            // deg >= 1 always
    deg_ns[n] = fmaxf(deg_ns[n], 1.0f);    // jnp.clip(..., 1.0)
  }
}

// ---------------------------------------------------------------------------
// Weight prep: all GEMM B-operands converted to f16 and transposed to [n][k]
// so a WMMA B-fragment is one contiguous, 32B-aligned v16h load per lane.
// Layout in wt: [3][128][128] conv | [64][128] soft_w1 | [64][128] hard_w1[:128]
// ---------------------------------------------------------------------------
__global__ __launch_bounds__(256) void prep_w(const float* __restrict__ w0,
                                              const float* __restrict__ w1,
                                              const float* __restrict__ w2,
                                              const float* __restrict__ sw1,
                                              const float* __restrict__ hw1,
                                              _Float16* __restrict__ wt) {
  int t = blockIdx.x * 256 + threadIdx.x;
  const int CONV = 3 * HH * HH;
  if (t < CONV) {
    int l = t / (HH * HH), r = t % (HH * HH);
    int k = r / HH, nI = r % HH;
    const float* w = (l == 0) ? w0 : ((l == 1) ? w1 : w2);
    wt[l * HH * HH + nI * HH + k] = (_Float16)w[k * HH + nI];
  } else if (t < CONV + 2 * CH * HH) {
    int r = t - CONV;
    int which = r / (CH * HH);         // 0 = soft, 1 = hard
    int rr = r % (CH * HH);
    int j = rr / HH, k = rr % HH;      // [j][k] <- w1[k][j]
    const float* w = which ? hw1 : sw1;
    wt[CONV + which * CH * HH + j * HH + k] = (_Float16)w[k * CH + j];
  }
}

// ---------------------------------------------------------------------------
// hw = h @ W  via v_wmma_f32_16x16x32_f16.
// Block = 8 waves: wave w owns the 16-col tile n0 = 16*w; block owns 16 rows.
// ---------------------------------------------------------------------------
__global__ __launch_bounds__(256) void gemm_wmma(const float* __restrict__ A,
                                                 const _Float16* __restrict__ Bt,
                                                 float* __restrict__ C) {
  const int lane = threadIdx.x & 31;
  const int wave = threadIdx.x >> 5;
  const int m0   = blockIdx.x * 16;
  const int n0   = wave * 16;
  const int row  = m0 + (lane & 15);
  const int col  = n0 + (lane & 15);
  const int hi   = (lane & 16) ? 1 : 0;
  const int kA1  = hi ? 16 : 0;
  const int kA2  = hi ? 24 : 8;
  const int kB   = hi ? 16 : 0;
  const float*    rp = A  + (size_t)row * HH;
  const _Float16* wp = Bt + (size_t)col * HH;

  v8f acc = {};
#pragma unroll
  for (int kb = 0; kb < HH; kb += 32) {
    float4 u0 = *(const float4*)(rp + kb + kA1);
    float4 u1 = *(const float4*)(rp + kb + kA1 + 4);
    float4 u2 = *(const float4*)(rp + kb + kA2);
    float4 u3 = *(const float4*)(rp + kb + kA2 + 4);
    v16h a;
    a[0]  = (_Float16)u0.x; a[1]  = (_Float16)u0.y;
    a[2]  = (_Float16)u0.z; a[3]  = (_Float16)u0.w;
    a[4]  = (_Float16)u1.x; a[5]  = (_Float16)u1.y;
    a[6]  = (_Float16)u1.z; a[7]  = (_Float16)u1.w;
    a[8]  = (_Float16)u2.x; a[9]  = (_Float16)u2.y;
    a[10] = (_Float16)u2.z; a[11] = (_Float16)u2.w;
    a[12] = (_Float16)u3.x; a[13] = (_Float16)u3.y;
    a[14] = (_Float16)u3.z; a[15] = (_Float16)u3.w;
    v16h b = *(const v16h*)(wp + kb + kB);
    acc = __builtin_amdgcn_wmma_f32_16x16x32_f16(false, a, false, b,
                                                 (short)0, acc, false, false);
  }
  const int mb = m0 + (hi ? 8 : 0);
#pragma unroll
  for (int r = 0; r < 8; ++r) C[(size_t)(mb + r) * HH + col] = acc[r];
}

// ---------------------------------------------------------------------------
// agg[n][c] = bias[c] + dinv[n]^2 * hw[n][c]   (self-loop term of gcn_norm)
// ---------------------------------------------------------------------------
__global__ __launch_bounds__(256) void agg_init(const float* __restrict__ hw,
                                                const float* __restrict__ bias,
                                                const float* __restrict__ dinv,
                                                float* __restrict__ agg) {
  long t = (long)blockIdx.x * 256 + threadIdx.x;
  if (t < (long)NN * HH) {
    int n = (int)(t / HH), c = (int)(t % HH);
    float dn = dinv[n];
    agg[t] = bias[c] + dn * dn * hw[t];
  }
}

// One wave per edge; lane owns 4 channels (float4 gather, 4 f32 atomics).
// Working set (hw = 25.6 MB) is L2-resident on the 192 MB L2.
__global__ __launch_bounds__(256) void agg_edges(const int* __restrict__ src,
                                                 const int* __restrict__ dst,
                                                 const float* __restrict__ dinv,
                                                 const float* __restrict__ hw,
                                                 float* __restrict__ agg) {
  const int lane = threadIdx.x & 31;
  const int e = blockIdx.x * 8 + (threadIdx.x >> 5);
  if (e >= EE) return;
  int s = src[e], d = dst[e];
  float nrm = dinv[s] * dinv[d];
  float4 hv = *((const float4*)(hw + (size_t)s * HH) + lane);
  float* ap = agg + (size_t)d * HH + lane * 4;
  unsafeAtomicAdd(ap + 0, nrm * hv.x);
  unsafeAtomicAdd(ap + 1, nrm * hv.y);
  unsafeAtomicAdd(ap + 2, nrm * hv.z);
  unsafeAtomicAdd(ap + 3, nrm * hv.w);
}

// Exact GELU (approximate=False), in place.
__global__ __launch_bounds__(256) void gelu_k(float* __restrict__ buf) {
  long t = (long)blockIdx.x * 256 + threadIdx.x;
  if (t < (long)NN * HH) {
    float v = buf[t];
    buf[t] = 0.5f * v * (1.0f + erff(v * 0.70710678118654752440f));
  }
}

__global__ __launch_bounds__(256) void ready_edges(const int* __restrict__ src,
                                                   const int* __restrict__ dst,
                                                   const float* __restrict__ soft_ex,
                                                   float* __restrict__ ready_sum) {
  int e = blockIdx.x * 256 + threadIdx.x;
  if (e < EE) {
    int s = src[e], d = dst[e];
    if (s != d) unsafeAtomicAdd(&ready_sum[s], soft_ex[d]);
  }
}

// ---------------------------------------------------------------------------
// Shared stage-1 helper: wave computes relu(h[strip] @ W1 + C0) for its
// 16-node strip into LDS. C0 (bias / bias + readiness rank-1 term) is folded
// into the WMMA C initializer. 16 WMMAs per wave.
// ---------------------------------------------------------------------------
__device__ __forceinline__ void head_stage1(const float* __restrict__ h,
                                            const _Float16* __restrict__ Wt,
                                            const float* __restrict__ b1,
                                            const float* __restrict__ hrow, // or nullptr
                                            const float* __restrict__ rd8,  // per-row readiness (8) or nullptr
                                            float* __restrict__ lds,
                                            int m0, int wave, int lane) {
  const int hi  = (lane & 16) ? 1 : 0;
  const int kA1 = hi ? 16 : 0;
  const int kA2 = hi ? 24 : 8;
  const int kB  = hi ? 16 : 0;
  const int row = m0 + wave * 16 + (lane & 15);
  const float* rp = h + (size_t)row * HH;

  // Hoist the 4 A-fragments (K=128) across all 4 N-tiles.
  v16h afr[4];
#pragma unroll
  for (int i = 0; i < 4; ++i) {
    const int kb = i * 32;
    float4 u0 = *(const float4*)(rp + kb + kA1);
    float4 u1 = *(const float4*)(rp + kb + kA1 + 4);
    float4 u2 = *(const float4*)(rp + kb + kA2);
    float4 u3 = *(const float4*)(rp + kb + kA2 + 4);
    v16h a;
    a[0]  = (_Float16)u0.x; a[1]  = (_Float16)u0.y;
    a[2]  = (_Float16)u0.z; a[3]  = (_Float16)u0.w;
    a[4]  = (_Float16)u1.x; a[5]  = (_Float16)u1.y;
    a[6]  = (_Float16)u1.z; a[7]  = (_Float16)u1.w;
    a[8]  = (_Float16)u2.x; a[9]  = (_Float16)u2.y;
    a[10] = (_Float16)u2.z; a[11] = (_Float16)u2.w;
    a[12] = (_Float16)u3.x; a[13] = (_Float16)u3.y;
    a[14] = (_Float16)u3.z; a[15] = (_Float16)u3.w;
    afr[i] = a;
  }

#pragma unroll
  for (int nt = 0; nt < 4; ++nt) {
    const int j = nt * 16 + (lane & 15);
    const _Float16* wp = Wt + (size_t)j * HH;
    v8f acc;
    float bj = b1[j];
    float hj = hrow ? hrow[j] : 0.0f;
#pragma unroll
    for (int r = 0; r < 8; ++r)
      acc[r] = rd8 ? (bj + rd8[r] * hj) : bj;
#pragma unroll
    for (int i = 0; i < 4; ++i) {
      v16h b = *(const v16h*)(wp + i * 32 + kB);
      acc = __builtin_amdgcn_wmma_f32_16x16x32_f16(false, afr[i], false, b,
                                                   (short)0, acc, false, false);
    }
    const int rbase = wave * 16 + (hi ? 8 : 0);
#pragma unroll
    for (int r = 0; r < 8; ++r)
      lds[(rbase + r) * LDS_STRIDE + j] = fmaxf(acc[r], 0.0f);
  }
}

// ---------------------------------------------------------------------------
// Soft-exit head (WMMA). 1 block = 128 nodes. Temperature head elided:
// dividing logits by temp>0 before softmax is order-preserving, so it can
// never change a decision.
// ---------------------------------------------------------------------------
__global__ __launch_bounds__(256) void soft_head(const float* __restrict__ h,
                                                 const _Float16* __restrict__ Wt,
                                                 const float* __restrict__ b1,
                                                 const float* __restrict__ W2,
                                                 const float* __restrict__ b2,
                                                 const float* __restrict__ gno,
                                                 float* __restrict__ soft_ex,
                                                 float* __restrict__ ready_sum) {
  __shared__ float lds[128 * LDS_STRIDE];
  const int lane = threadIdx.x & 31;
  const int wave = threadIdx.x >> 5;
  const int m0   = blockIdx.x * 128;

  head_stage1(h, Wt, b1, nullptr, nullptr, lds, m0, wave, lane);
  __syncthreads();

  // Stage 2: 2 lanes per node: node = lane>>1, class c = lane&1.
  const int t = lane >> 1, c = lane & 1;
  const int node = m0 + wave * 16 + t;
  const float* hid = lds + (wave * 16 + t) * LDS_STRIDE;
  float acc = b2[c];
#pragma unroll 8
  for (int k = 0; k < CH; ++k) acc = fmaf(hid[k], W2[k * 2 + c], acc);
  float other = __shfl_xor(acc, 1, 32);    // class-paired logit
  if (c == 0 && node < NN) {
    float g0 = gno[node * 2], g1 = gno[node * 2 + 1];
    if ((other + g1) > (acc + g0)) soft_ex[node] = 1.0f;  // |= decision
    ready_sum[node] = 0.0f;                               // pre-zero readiness
  }
}

// ---------------------------------------------------------------------------
// Hard-exit head (WMMA, K=129). Readiness row folded into the C initializer
// as a rank-1 term. Decisions, z accumulation, exit_layers, active count.
// ---------------------------------------------------------------------------
__global__ __launch_bounds__(256) void hard_head(const float* __restrict__ h,
                                                 const _Float16* __restrict__ Wt,
                                                 const float* __restrict__ b1,
                                                 const float* __restrict__ hrow, // hard_w1 row 128 (f32, 64)
                                                 const float* __restrict__ W2,
                                                 const float* __restrict__ b2,
                                                 const float* __restrict__ gno,
                                                 const float* __restrict__ soft_ex,
                                                 const float* __restrict__ ready_sum,
                                                 const float* __restrict__ deg_ns,
                                                 int* __restrict__ hard_ex,
                                                 int* __restrict__ exit_l,
                                                 int* __restrict__ active,
                                                 float* __restrict__ z,
                                                 int li) {
  __shared__ float lds[128 * LDS_STRIDE];
  const int lane = threadIdx.x & 31;
  const int wave = threadIdx.x >> 5;
  const int m0   = blockIdx.x * 128;
  const int hi   = (lane & 16) ? 1 : 0;

  // Per-C-row readiness for the rank-1 correction (rows m = base + r + 8*hi).
  float rd8[8];
#pragma unroll
  for (int r = 0; r < 8; ++r) {
    int m = m0 + wave * 16 + (hi ? 8 : 0) + r;
    rd8[r] = (m < NN) ? (ready_sum[m] / deg_ns[m]) : 0.0f;
  }

  head_stage1(h, Wt, b1, hrow, rd8, lds, m0, wave, lane);
  __syncthreads();

  const int t = lane >> 1, c = lane & 1;
  const int node = m0 + wave * 16 + t;
  const float* hid = lds + (wave * 16 + t) * LDS_STRIDE;
  float acc = b2[c];
#pragma unroll 8
  for (int k = 0; k < CH; ++k) acc = fmaf(hid[k], W2[k * 2 + c], acc);
  float other = __shfl_xor(acc, 1, 32);

  bool newly = false, countme = false;
  if (c == 0 && node < NN) {
    bool  se = soft_ex[node] > 0.5f;
    float e0 = se ? acc   : 1.0e6f;
    float e1 = se ? other : 0.0f;
    bool dec = (e1 + gno[node * 2 + 1]) > (e0 + gno[node * 2]);
    int  he  = hard_ex[node];
    countme  = (he == 0);
    newly    = dec && (he == 0);
    if (newly) { exit_l[node] = li; hard_ex[node] = 1; }
  }
  // One atomic per wave for the active count.
  unsigned long long cb = __ballot(countme);
  if (lane == 0 && cb) atomicAdd(&active[li], __popcll(cb));
  // Cooperative z += h for newly-exited nodes (all 32 lanes, float4 each).
  unsigned long long bal = __ballot(newly);
  for (int t2 = 0; t2 < 16; ++t2) {
    if (bal & (1ull << (2 * t2))) {
      int nd = m0 + wave * 16 + t2;
      float4* zp = (float4*)(z + (size_t)nd * HH) + lane;
      float4  hv = *((const float4*)(h + (size_t)nd * HH) + lane);
      float4  zv = *zp;
      zv.x += hv.x; zv.y += hv.y; zv.z += hv.z; zv.w += hv.w;
      *zp = zv;
    }
  }
}

// Residual for never-exited nodes + copy int outputs into d_out (bitwise).
__global__ __launch_bounds__(256) void finalize(const float* __restrict__ h,
                                                const int* __restrict__ hard_ex,
                                                const int* __restrict__ exit_l,
                                                const int* __restrict__ active,
                                                float* __restrict__ out) {
  const int lane = threadIdx.x & 31;
  const int n = blockIdx.x * 8 + (threadIdx.x >> 5);
  if (n < NN) {
    if (hard_ex[n] == 0) {
      float4* zp = (float4*)(out + (size_t)n * HH) + lane;
      float4  hv = *((const float4*)(h + (size_t)n * HH) + lane);
      float4  zv = *zp;
      zv.x += hv.x; zv.y += hv.y; zv.z += hv.z; zv.w += hv.w;
      *zp = zv;
    }
    if (lane == 0) ((int*)out)[(size_t)NN * HH + n] = exit_l[n];
  }
  if (blockIdx.x == 0 && threadIdx.x < LL)
    ((int*)out)[(size_t)NN * HH + NN + threadIdx.x] = active[threadIdx.x];
}

// ---------------------------------------------------------------------------
extern "C" void kernel_launch(void* const* d_in, const int* in_sizes, int n_in,
                              void* d_out, int out_size, void* d_ws, size_t ws_size,
                              hipStream_t stream) {
  const float* x        = (const float*)d_in[0];
  const int*   src      = (const int*)d_in[1];            // edge_index[0]
  const int*   dst      = ((const int*)d_in[1]) + EE;     // edge_index[1]
  const float* gsoft    = (const float*)d_in[2];          // [L,N,2]
  const float* ghard    = (const float*)d_in[3];          // [L,N,2]
  const float* conv_w[3] = { (const float*)d_in[4], (const float*)d_in[6],
                             (const float*)d_in[8] };
  const float* conv_b[3] = { (const float*)d_in[5], (const float*)d_in[7],
                             (const float*)d_in[9] };
  const float* soft_w1 = (const float*)d_in[10];
  const float* soft_b1 = (const float*)d_in[11];
  const float* soft_w2 = (const float*)d_in[12];
  const float* soft_b2 = (const float*)d_in[13];
  const float* hard_w1 = (const float*)d_in[14];
  const float* hard_b1 = (const float*)d_in[15];
  const float* hard_w2 = (const float*)d_in[16];
  const float* hard_b2 = (const float*)d_in[17];
  // d_in[18] = temp_w: provably irrelevant to outputs (order-preserving) -> unused

  float* out = (float*)d_out;

  // workspace layout
  float* ws      = (float*)d_ws;
  float* hw      = ws;                              // N*H
  float* hbuf    = hw + (size_t)NN * HH;            // N*H (agg/h, in place)
  float* deg     = hbuf + (size_t)NN * HH;
  float* dinv    = deg + NN;
  float* deg_ns  = dinv + NN;
  float* ready   = deg_ns + NN;
  float* soft_ex = ready + NN;
  int*   hard_ex = (int*)(soft_ex + NN);
  int*   exit_l  = hard_ex + NN;
  int*   active  = exit_l + NN;
  _Float16* wt      = (_Float16*)(active + 8);      // [3][128][128] conv f16^T
  _Float16* wt_soft = wt + 3 * HH * HH;             // [64][128]
  _Float16* wt_hard = wt_soft + CH * HH;            // [64][128] (rows 0..127)
  const float* hrow = hard_w1 + HH * CH;            // hard_w1 row 128 (readiness)

  const dim3 B(256);
  const int gElem  = (NN * HH + 255) / 256;   // 25000
  const int gEdge  = (EE + 255) / 256;        // 3125
  const int gEdgeW = EE / 8;                  // 100000 (wave per edge)
  const int gNode  = (NN + 255) / 256;        // 196
  const int gWave  = NN / 8;                  // 6250  (wave per node)
  const int gGemm  = NN / 16;                 // 3125  (16-row tiles)
  const int gHead  = (NN + 127) / 128;        // 391   (128 nodes per block)
  const int gPrep  = (3 * HH * HH + 2 * CH * HH + 255) / 256;

  init_all<<<gElem, B, 0, stream>>>(out, soft_ex, hard_ex, exit_l, active, deg, deg_ns);
  deg_edges<<<gEdge, B, 0, stream>>>(src, dst, deg, deg_ns);
  dinv_fix<<<gNode, B, 0, stream>>>(deg, dinv, deg_ns);
  prep_w<<<gPrep, B, 0, stream>>>(conv_w[0], conv_w[1], conv_w[2], soft_w1, hard_w1, wt);

  for (int li = 0; li < LL; ++li) {
    const float* hin = (li == 0) ? x : hbuf;
    gemm_wmma<<<gGemm, B, 0, stream>>>(hin, wt + (size_t)li * HH * HH, hw);
    agg_init <<<gElem, B, 0, stream>>>(hw, conv_b[li], dinv, hbuf);
    agg_edges<<<gEdgeW, B, 0, stream>>>(src, dst, dinv, hw, hbuf);
    if (li < LL - 1) gelu_k<<<gElem, B, 0, stream>>>(hbuf);
    soft_head<<<gHead, B, 0, stream>>>(hbuf, wt_soft, soft_b1, soft_w2, soft_b2,
                                       gsoft + (size_t)li * NN * 2, soft_ex, ready);
    ready_edges<<<gEdge, B, 0, stream>>>(src, dst, soft_ex, ready);
    hard_head<<<gHead, B, 0, stream>>>(hbuf, wt_hard, hard_b1, hrow, hard_w2, hard_b2,
                                       ghard + (size_t)li * NN * 2, soft_ex, ready,
                                       deg_ns, hard_ex, exit_l, active, out, li);
  }
  finalize<<<gWave, B, 0, stream>>>(hbuf, hard_ex, exit_l, active, out);
}
